// SpeechEncVol7_58884001628648
// MI455X (gfx1250) — matrix operand order repaired and checked by reading, not compiled, not run on hardware
//
#include <hip/hip_runtime.h>
#include <hip/hip_bf16.h>

typedef __bf16 v16bf __attribute__((ext_vector_type(16)));
typedef float  v8f   __attribute__((ext_vector_type(8)));
typedef unsigned int v4u __attribute__((ext_vector_type(4)));

union V16BF { v4u u[2]; v16bf v; };

__device__ __forceinline__ v8f wmma_bf16(const V16BF& a, const V16BF& b, v8f c) {
    return __builtin_amdgcn_wmma_f32_16x16x32_bf16(false, a.v, false, b.v, (short)0, c, false, false);
}

// ---------------- constants ----------------
#define NB   32          // batch
#define CIN  13
#define LIN  4096
#define C1   128         // conv1 out channels
#define L1   4094        // conv1 out length
#define HH   256         // hidden
#define G3   768         // 3*H
#define LT   2046        // subsampled length
#define MROWS (NB*LT)    // 65472 GEMM rows
#define MTILES (MROWS/16) // 4092

// ---------------- prep: f32 (R,C) -> bf16 (R,C) convert, same layout ----------------
__global__ __launch_bounds__(256) void k_transpose_bf16(const float* __restrict__ in,
                                                        __hip_bfloat16* __restrict__ out,
                                                        int R, int C) {
    int idx = blockIdx.x * 256 + threadIdx.x;
    if (idx >= R * C) return;
    int r = idx / C, c = idx % C;
    out[(size_t)r * C + c] = __float2bfloat16(in[(size_t)r * C + c]);
}

// W2 (256,128,3) f32 -> W2T[o][kk*128+c] bf16  (K=384, N=256)
__global__ __launch_bounds__(256) void k_w2t(const float* __restrict__ W2,
                                             __hip_bfloat16* __restrict__ W2T) {
    int idx = blockIdx.x * 256 + threadIdx.x;   // 256*128*3 = 98304
    if (idx >= 256 * 128 * 3) return;
    int o = idx / 384;
    int r = idx % 384;
    int c = r / 3, kk = r % 3;
    W2T[(size_t)o * 384 + kk * 128 + c] = __float2bfloat16(W2[idx]);
}

// ---------------- conv1 (direct, VALU) -> y1t[n][p][c] bf16 ----------------
__global__ __launch_bounds__(128) void k_conv1(const float* __restrict__ X,
                                               const float* __restrict__ W1,
                                               const float* __restrict__ b1,
                                               __hip_bfloat16* __restrict__ y1) {
    int p = blockIdx.x;          // 0..4093
    int n = blockIdx.y;          // 0..31
    int c = threadIdx.x;         // 0..127
    __shared__ float xs[CIN * 3];
    if (threadIdx.x < CIN * 3) {
        int ic = threadIdx.x / 3, kk = threadIdx.x % 3;
        xs[threadIdx.x] = X[((size_t)n * CIN + ic) * LIN + p + kk];
    }
    __syncthreads();
    float acc = b1[c];
    const float* w = W1 + (size_t)c * (CIN * 3);
    #pragma unroll
    for (int q = 0; q < CIN * 3; ++q) acc = fmaf(w[q], xs[q], acc);
    y1[((size_t)n * L1 + p) * C1 + c] = __float2bfloat16(acc);
}

// ---------------- conv2 as WMMA GEMM (even positions only) -------------
// A rows = (n,t): A[row, kk*128+c] = y1[n][2t+kk][c]   K = 384
// B = W2T (K x 256).  Output -> x_bf[(t*32+n)*256 + o]  bf16
__global__ __launch_bounds__(256) void k_conv2(const __hip_bfloat16* __restrict__ y1,
                                               const __hip_bfloat16* __restrict__ w2t,
                                               const float* __restrict__ b2,
                                               __hip_bfloat16* __restrict__ xbf) {
    const int lane = threadIdx.x & 31;
    const int wave = threadIdx.x >> 5;
    const int mt = blockIdx.y * 8 + wave;
    if (mt >= MTILES) return;                       // wave-uniform
    const int jt = blockIdx.x;                      // 0..15
    const int m = lane & 15;
    const int hs = lane >> 4;
    const int row = mt * 16 + m;
    const int n = row / LT, t = row % LT;

    v8f acc = {};
    #pragma unroll
    for (int kt = 0; kt < 12; ++kt) {
        const int kk = (kt * 32) >> 7;
        const int c0 = ((kt * 32) & 127) + hs * 8;
        V16BF a;
        const v4u* ap = (const v4u*)(y1 + ((size_t)(n * L1 + 2 * t + kk)) * C1 + c0);
        a.u[0] = ap[0];
        a.u[1] = ap[2];                              // +16 channels
        V16BF b;
        const v4u* bp = (const v4u*)(w2t + (size_t)(jt * 16 + m) * 384 + kt * 32 + hs * 8);
        b.u[0] = bp[0]; b.u[1] = bp[2];
        acc = wmma_bf16(a, b, acc);
    }
    const int col = jt * 16 + m;
    const float bias = b2[col];
    #pragma unroll
    for (int v = 0; v < 8; ++v) {
        int rg = mt * 16 + hs * 8 + v;
        int nn = rg / LT, tt = rg % LT;
        xbf[((size_t)(tt * NB + nn)) * HH + col] = __float2bfloat16(acc[v] + bias);
    }
}

// ---------------- gi GEMM: gi = x @ WihT + bih  (both dirs) -------------
__global__ __launch_bounds__(256) void k_gi(const __hip_bfloat16* __restrict__ xbf,
                                            const __hip_bfloat16* __restrict__ wT_f,
                                            const __hip_bfloat16* __restrict__ wT_b,
                                            const float* __restrict__ bih_f,
                                            const float* __restrict__ bih_b,
                                            float* __restrict__ gi_f,
                                            float* __restrict__ gi_b) {
    const int dir = blockIdx.z;
    const __hip_bfloat16* w = dir ? wT_b : wT_f;
    const float* bih = dir ? bih_b : bih_f;
    float* gi = dir ? gi_b : gi_f;

    const int lane = threadIdx.x & 31;
    const int wave = threadIdx.x >> 5;
    const int mt = blockIdx.y * 8 + wave;
    if (mt >= MTILES) return;
    const int jt = blockIdx.x;                      // 0..47
    const int m = lane & 15;
    const int hs = lane >> 4;
    const int row = mt * 16 + m;

    v8f acc = {};
    #pragma unroll
    for (int kt = 0; kt < 8; ++kt) {
        V16BF a;
        const v4u* ap = (const v4u*)(xbf + (size_t)row * HH + kt * 32 + hs * 8);
        a.u[0] = ap[0];
        a.u[1] = ap[2];
        V16BF b;
        const v4u* bp = (const v4u*)(w + (size_t)(jt * 16 + m) * HH + kt * 32 + hs * 8);
        b.u[0] = bp[0]; b.u[1] = bp[2];
        acc = wmma_bf16(a, b, acc);
    }
    const int col = jt * 16 + m;
    const float bias = bih[col];
    #pragma unroll
    for (int v = 0; v < 8; ++v) {
        int rg = mt * 16 + hs * 8 + v;
        gi[(size_t)rg * G3 + col] = acc[v] + bias;
    }
}

// ---------------- persistent GRU scan: 1 block per direction -----------
// block = 1024 threads = 32 waves.  LDS: gh(96K f32) + h f32(32K) + h bf16(16K) + bhh(3K)
__global__ __launch_bounds__(1024) void k_gru_scan(const float* __restrict__ gi_f,
                                                   const float* __restrict__ gi_b,
                                                   const __hip_bfloat16* __restrict__ whhT_f,
                                                   const __hip_bfloat16* __restrict__ whhT_b,
                                                   const float* __restrict__ bhh_f,
                                                   const float* __restrict__ bhh_b,
                                                   float* __restrict__ out_f,
                                                   float* __restrict__ out_b) {
    const int dir = blockIdx.x;
    const float* gi = dir ? gi_b : gi_f;
    const __hip_bfloat16* whhT = dir ? whhT_b : whhT_f;
    const float* bhh = dir ? bhh_b : bhh_f;
    float* out = dir ? out_b : out_f;

    extern __shared__ char smem[];
    float* gh = (float*)smem;                                  //  98304 B
    float* hf = (float*)(smem + 98304);                        //  32768 B
    __hip_bfloat16* hb = (__hip_bfloat16*)(smem + 131072);     //  16384 B
    float* sb = (float*)(smem + 147456);                       //   3072 B

    const int tid = threadIdx.x;
    const int lane = tid & 31;
    const int wave = tid >> 5;
    const int mt = wave >> 4;             // 0..1
    const int nt0 = (wave & 15) * 3;      // 0,3,...,45
    const int hs = lane >> 4;
    const int m = lane & 15;
    const int arow = mt * 16 + m;

    for (int e = tid; e < NB * HH; e += 1024) { hf[e] = 0.f; hb[e] = __float2bfloat16(0.f); }
    for (int e = tid; e < G3; e += 1024) sb[e] = bhh[e];
    __syncthreads();

    const int ei = tid >> 5;              // batch row for elementwise (0..31)
    const int ej = (tid * 8) & 255;       // hidden col base

    for (int s = 0; s < LT; ++s) {
        const int l = dir ? (LT - 1 - s) : s;
        const float* git = gi + (size_t)l * (NB * G3);

        if (s + 1 < LT) {   // prefetch next step's gi slice (gfx1250 global_prefetch_b8)
            const int ln = dir ? (LT - 2 - s) : (s + 1);
            __builtin_prefetch(gi + (size_t)ln * (NB * G3) + tid * 24, 0, 1);
        }

        // phase 1: gh = h @ WhhT   (A from LDS, B streamed from L2)
        v8f acc0 = {}, acc1 = {}, acc2 = {};
        #pragma unroll
        for (int kt = 0; kt < 8; ++kt) {
            V16BF a;
            const v4u* ap = (const v4u*)(hb + arow * HH + kt * 32 + hs * 8);
            a.u[0] = ap[0];
            a.u[1] = ap[2];
            const v4u* bp = (const v4u*)(whhT + (size_t)(nt0 * 16 + m) * HH + kt * 32 + hs * 8);
            V16BF b0, b1, b2;
            b0.u[0] = bp[0]; b0.u[1] = bp[2];
            b1.u[0] = bp[512]; b1.u[1] = bp[514];
            b2.u[0] = bp[1024]; b2.u[1] = bp[1026];
            acc0 = wmma_bf16(a, b0, acc0);
            acc1 = wmma_bf16(a, b1, acc1);
            acc2 = wmma_bf16(a, b2, acc2);
        }

        // phase 2: scatter C tiles into LDS gh
        {
            const int crow0 = mt * 16 + hs * 8;
            #pragma unroll
            for (int v = 0; v < 8; ++v) {
                float* g = gh + (size_t)(crow0 + v) * G3 + m;
                g[(nt0 + 0) * 16] = acc0[v];
                g[(nt0 + 1) * 16] = acc1[v];
                g[(nt0 + 2) * 16] = acc2[v];
            }
        }
        __syncthreads();

        // phase 3: gates + state update; 8 hidden elems per thread
        {
            const float* gr = git + (size_t)ei * G3;
            float* orow = out + (size_t)(l * NB + ei) * HH;
            #pragma unroll
            for (int u = 0; u < 8; ++u) {
                const int j = ej + u;
                const float ghr = gh[(size_t)ei * G3 + j]       + sb[j];
                const float ghz = gh[(size_t)ei * G3 + 256 + j] + sb[256 + j];
                const float ghn = gh[(size_t)ei * G3 + 512 + j] + sb[512 + j];
                const float r = 1.f / (1.f + __expf(-(gr[j] + ghr)));
                const float z = 1.f / (1.f + __expf(-(gr[256 + j] + ghz)));
                const float nn = tanhf(gr[512 + j] + r * ghn);
                const float hv = (1.f - z) * nn + z * hf[ei * HH + j];
                orow[j] = hv;
                hf[ei * HH + j] = hv;
                hb[ei * HH + j] = __float2bfloat16(hv);
            }
        }
        __syncthreads();
    }
}

// ---------------- combine: relu(0.5*(out_f + rev_h(out_b))) + hT_b -----
__global__ __launch_bounds__(256) void k_combine(const float* __restrict__ out_f,
                                                 const float* __restrict__ out_b,
                                                 float* __restrict__ dst) {
    const size_t total = (size_t)LT * NB * HH;
    const size_t stride = (size_t)gridDim.x * blockDim.x;
    for (size_t e = (size_t)blockIdx.x * blockDim.x + threadIdx.x;
         e < total + NB * HH; e += stride) {
        if (e < total) {
            size_t rem = e & (size_t)(NB * HH - 1);   // 8192 = pow2
            size_t l = e >> 13;
            int i = (int)(rem >> 8);
            int j = (int)(rem & 255);
            float v = 0.5f * (out_f[e] + out_b[(l * NB + i) * (size_t)HH + (255 - j)]);
            dst[e] = v > 0.f ? v : 0.f;
        } else {
            size_t e2 = e - total;                    // hT_b = out_b at l=0
            dst[e] = out_b[e2];
        }
    }
}

// ---------------- host launch ----------------
extern "C" void kernel_launch(void* const* d_in, const int* in_sizes, int n_in,
                              void* d_out, int out_size, void* d_ws, size_t ws_size,
                              hipStream_t stream) {
    (void)in_sizes; (void)n_in; (void)out_size; (void)ws_size;
    const float* X     = (const float*)d_in[0];
    const float* W1    = (const float*)d_in[1];
    const float* b1    = (const float*)d_in[2];
    const float* W2    = (const float*)d_in[3];
    const float* b2    = (const float*)d_in[4];
    const float* Wih_f = (const float*)d_in[5];
    const float* Whh_f = (const float*)d_in[6];
    const float* bih_f = (const float*)d_in[7];
    const float* bhh_f = (const float*)d_in[8];
    const float* Wih_b = (const float*)d_in[9];
    const float* Whh_b = (const float*)d_in[10];
    const float* bih_b = (const float*)d_in[11];
    const float* bhh_b = (const float*)d_in[12];

    char* ws = (char*)d_ws;
    size_t off = 0;
    __hip_bfloat16* y1    = (__hip_bfloat16*)(ws + off); off += (size_t)NB * L1 * C1 * 2;    // 33,538,048
    __hip_bfloat16* xbf   = (__hip_bfloat16*)(ws + off); off += (size_t)MROWS * HH * 2;      // 33,521,664
    __hip_bfloat16* W2T   = (__hip_bfloat16*)(ws + off); off += (size_t)384 * 256 * 2;
    __hip_bfloat16* WihTf = (__hip_bfloat16*)(ws + off); off += (size_t)HH * G3 * 2;
    __hip_bfloat16* WihTb = (__hip_bfloat16*)(ws + off); off += (size_t)HH * G3 * 2;
    __hip_bfloat16* WhhTf = (__hip_bfloat16*)(ws + off); off += (size_t)HH * G3 * 2;
    __hip_bfloat16* WhhTb = (__hip_bfloat16*)(ws + off); off += (size_t)HH * G3 * 2;
    float* gi_f  = (float*)(ws + off); off += (size_t)MROWS * G3 * 4;                        // 201 MB
    float* gi_b  = (float*)(ws + off); off += (size_t)MROWS * G3 * 4;
    float* out_f = (float*)(ws + off); off += (size_t)MROWS * HH * 4;                        // 67 MB
    float* out_b = (float*)(ws + off); off += (size_t)MROWS * HH * 4;

    // weight prep (bf16 K-major)
    k_transpose_bf16<<<dim3((G3 * HH + 255) / 256), dim3(256), 0, stream>>>(Wih_f, WihTf, G3, HH);
    k_transpose_bf16<<<dim3((G3 * HH + 255) / 256), dim3(256), 0, stream>>>(Wih_b, WihTb, G3, HH);
    k_transpose_bf16<<<dim3((G3 * HH + 255) / 256), dim3(256), 0, stream>>>(Whh_f, WhhTf, G3, HH);
    k_transpose_bf16<<<dim3((G3 * HH + 255) / 256), dim3(256), 0, stream>>>(Whh_b, WhhTb, G3, HH);
    k_w2t<<<dim3((256 * 128 * 3 + 255) / 256), dim3(256), 0, stream>>>(W2, W2T);

    // conv stack
    k_conv1<<<dim3(L1, NB), dim3(128), 0, stream>>>(X, W1, b1, y1);
    k_conv2<<<dim3(16, (MTILES + 7) / 8), dim3(256), 0, stream>>>(y1, W2T, b2, xbf);

    // input projections (both dirs)
    k_gi<<<dim3(48, (MTILES + 7) / 8, 2), dim3(256), 0, stream>>>(xbf, WihTf, WihTb,
                                                                  bih_f, bih_b, gi_f, gi_b);

    // sequential bidirectional scan: 2 persistent workgroups, 150528 B dynamic LDS
    k_gru_scan<<<dim3(2), dim3(1024), 150528, stream>>>(gi_f, gi_b, WhhTf, WhhTb,
                                                        bhh_f, bhh_b, out_f, out_b);

    // final combine + hidden state
    k_combine<<<dim3(4096), dim3(256), 0, stream>>>(out_f, out_b, (float*)d_out);
}
